// SimpleTransformer_3985729651336
// MI455X (gfx1250) — matrix operand order repaired and checked by reading, not compiled, not run on hardware
//
#include <hip/hip_runtime.h>
#include <hip/hip_bf16.h>

// ---------------- problem constants ----------------
#define B_IMG   64
#define NTOK    196      // 14*14 patches
#define NPAD    256      // padded token count (multiple of 128)
#define DIMV    1024
#define PD      768      // 3*16*16
#define NHEAD   8
#define DHEAD   128
#define DEPTH   6
#define LNEPS   1e-5f
#define NCLS    1000

typedef __attribute__((ext_vector_type(16))) __bf16 v16bf;
typedef __attribute__((ext_vector_type(8)))  __bf16 v8bf;
typedef __attribute__((ext_vector_type(8)))  float  v8f;
typedef __attribute__((ext_vector_type(4)))  int    v4i;

union FragU { v16bf v; v8bf h[2]; };

// ---------------- async global->LDS copy (gfx1250 path w/ fallback) ----------------
#if defined(__has_builtin)
#  if __has_builtin(__builtin_amdgcn_global_load_async_to_lds_b128)
#    define HAVE_ASYNC_LDS 1
#  endif
#endif
#ifndef HAVE_ASYNC_LDS
#  define HAVE_ASYNC_LDS 0
#endif

__device__ inline void cp16_g2l(const __bf16* gp, __bf16* lp) {
#if HAVE_ASYNC_LDS
  __builtin_amdgcn_global_load_async_to_lds_b128((v4i*)gp, (v4i*)lp, 0, 0);
#else
  *(v8bf*)lp = *(const v8bf*)gp;
#endif
}

__device__ inline void stage_wait(bool more_in_flight) {
#if HAVE_ASYNC_LDS
  if (more_in_flight) asm volatile("s_wait_asynccnt 4" ::: "memory");
  else                asm volatile("s_wait_asynccnt 0" ::: "memory");
#else
  (void)more_in_flight;
#endif
  __syncthreads();
}

// ---------------- GEMM tile geometry ----------------
#define BM 128
#define BN 128
#define BK 32
#define LSTR 40   // LDS row stride in bf16 elements (80B, 16B aligned, bank-skewed)

// ---- WMMA fragment loaders from LDS (bf16, 16x16x32 shapes) ----
// A (16x32, MxK): lane l holds row m = l&15, khalf = l>>4.
// element e: K = (e>>3)*16 + khalf*8 + (e&7)  -> two 8-contiguous groups.
__device__ inline v16bf lds_fragA(const __bf16* sa, int r0, int lane) {
  const __bf16* p = sa + (r0 + (lane & 15)) * LSTR + ((lane >> 4) * 8);
  FragU f;
  f.h[0] = *(const v8bf*)p;
  f.h[1] = *(const v8bf*)(p + 16);
  return f.v;
}
// B (32x16, KxN) from K-contiguous (N-major) LDS tile:
// lane l holds col n = l&15, khalf = l>>4; element e: K = (e>>3)*16 + khalf*8 + (e&7) (same as A).
__device__ inline v16bf lds_fragB(const __bf16* sb, int c0, int lane) {
  const __bf16* p = sb + (c0 + (lane & 15)) * LSTR + ((lane >> 4) * 8);
  FragU f;
  f.h[0] = *(const v8bf*)p;
  f.h[1] = *(const v8bf*)(p + 16);
  return f.v;
}

// stage one BMxBK A tile and BNxBK B tile into LDS (4 x b128 per thread)
__device__ inline void stage_tile(const __bf16* __restrict__ A, int lda, int row0,
                                  const __bf16* __restrict__ Bt, int ldb, int col0,
                                  int kb, __bf16* sa, __bf16* sb) {
  int t = threadIdx.x;
  #pragma unroll
  for (int i = 0; i < 2; i++) {           // A: 128x32 = 512 chunks of 16B
    int idx = t + i * 256;
    int r = idx >> 2, c = (idx & 3) * 8;
    cp16_g2l(A + (size_t)(row0 + r) * lda + kb + c, sa + r * LSTR + c);
  }
  #pragma unroll
  for (int i = 0; i < 2; i++) {           // B: 128x32 = 512 chunks of 16B
    int idx = t + i * 256;
    int r = idx >> 2, c = (idx & 3) * 8;
    cp16_g2l(Bt + (size_t)(col0 + r) * ldb + kb + c, sb + r * LSTR + c);
  }
}

#define OUT_F32    0
#define OUT_BF16   1
#define OUT_BF16_T 2   // store C transposed (col-major) as bf16

// C[M,N] = alpha * A[M,K] @ Bt[N,K]^T  (+bias[n]) (+resid[m,n]),  per-batch strides.
// Block = 256 threads = 8 waves (4x2 grid); block tile 128x128; wave tile 32x64 (2x4 WMMA).
// Double-buffered async global->LDS staging, ds_load_b128 fragment reads.
__global__ __launch_bounds__(256)
void wmma_gemm_bf16(const __bf16* __restrict__ A, long long sAb, int lda,
                    const __bf16* __restrict__ Bt, long long sBb, int ldb,
                    void* __restrict__ Cv, long long sCb, int ldc,
                    const __bf16* __restrict__ R, long long sRb, int ldr,
                    const float* __restrict__ bias,
                    int K, float alpha, int outMode) {
  __shared__ __bf16 sA[2][BM * LSTR];
  __shared__ __bf16 sB[2][BN * LSTR];

  int z = blockIdx.z;
  A  += (size_t)z * sAb;
  Bt += (size_t)z * sBb;
  const __bf16* Rb = R ? (R + (size_t)z * sRb) : nullptr;

  int lane = threadIdx.x & 31;
  int w    = threadIdx.x >> 5;
  int trow = (w >> 1) * 32;           // wave row inside block tile (0,32,64,96)
  int tcol = (w & 1) * 64;            // wave col inside block tile (0,64)
  int row0 = blockIdx.y * BM;
  int col0 = blockIdx.x * BN;

  v8f acc[2][4] = {};

  stage_tile(A, lda, row0, Bt, ldb, col0, 0, sA[0], sB[0]);
  int nk = K / BK;
  for (int ki = 0; ki < nk; ki++) {
    int cur = ki & 1;
    bool more = (ki + 1) < nk;
    if (more)
      stage_tile(A, lda, row0, Bt, ldb, col0, (ki + 1) * BK, sA[cur ^ 1], sB[cur ^ 1]);
    stage_wait(more);                 // current buffer complete on all waves

    const __bf16* sa = sA[cur];
    const __bf16* sb = sB[cur];
    v16bf a0 = lds_fragA(sa, trow,      lane);
    v16bf a1 = lds_fragA(sa, trow + 16, lane);
    v16bf b0 = lds_fragB(sb, tcol,      lane);
    v16bf b1 = lds_fragB(sb, tcol + 16, lane);
    v16bf b2 = lds_fragB(sb, tcol + 32, lane);
    v16bf b3 = lds_fragB(sb, tcol + 48, lane);
    acc[0][0] = __builtin_amdgcn_wmma_f32_16x16x32_bf16(false, a0, false, b0, (short)0, acc[0][0], false, false);
    acc[0][1] = __builtin_amdgcn_wmma_f32_16x16x32_bf16(false, a0, false, b1, (short)0, acc[0][1], false, false);
    acc[0][2] = __builtin_amdgcn_wmma_f32_16x16x32_bf16(false, a0, false, b2, (short)0, acc[0][2], false, false);
    acc[0][3] = __builtin_amdgcn_wmma_f32_16x16x32_bf16(false, a0, false, b3, (short)0, acc[0][3], false, false);
    acc[1][0] = __builtin_amdgcn_wmma_f32_16x16x32_bf16(false, a1, false, b0, (short)0, acc[1][0], false, false);
    acc[1][1] = __builtin_amdgcn_wmma_f32_16x16x32_bf16(false, a1, false, b1, (short)0, acc[1][1], false, false);
    acc[1][2] = __builtin_amdgcn_wmma_f32_16x16x32_bf16(false, a1, false, b2, (short)0, acc[1][2], false, false);
    acc[1][3] = __builtin_amdgcn_wmma_f32_16x16x32_bf16(false, a1, false, b3, (short)0, acc[1][3], false, false);
    __syncthreads();                  // reads done before buffer reuse
  }

  float*  Cf = (float*)Cv;
  __bf16* Ch = (__bf16*)Cv;
  size_t  cb = (size_t)z * sCb;
  #pragma unroll
  for (int i = 0; i < 2; i++)
    #pragma unroll
    for (int j = 0; j < 4; j++) {
      int n  = col0 + tcol + j * 16 + (lane & 15);
      int mb = row0 + trow + i * 16 + (lane >> 4) * 8;
      #pragma unroll
      for (int e = 0; e < 8; e++) {
        int m = mb + e;
        float v = acc[i][j][e] * alpha;
        if (bias) v += bias[n];
        if (Rb)   v += (float)Rb[(size_t)m * ldr + n];
        if (outMode == OUT_F32)       Cf[cb + (size_t)m * ldc + n] = v;
        else if (outMode == OUT_BF16) Ch[cb + (size_t)m * ldc + n] = (__bf16)v;
        else                          Ch[cb + (size_t)n * ldc + m] = (__bf16)v;
      }
    }
}

// ---------------- reductions ----------------
__device__ inline float block_sum(float v, float* sbuf) {
  int t = threadIdx.x;
  sbuf[t] = v; __syncthreads();
  for (int s = 128; s > 0; s >>= 1) { if (t < s) sbuf[t] += sbuf[t + s]; __syncthreads(); }
  float r = sbuf[0]; __syncthreads();
  return r;
}
__device__ inline float block_max(float v, float* sbuf) {
  int t = threadIdx.x;
  sbuf[t] = v; __syncthreads();
  for (int s = 128; s > 0; s >>= 1) { if (t < s) sbuf[t] = fmaxf(sbuf[t], sbuf[t + s]); __syncthreads(); }
  float r = sbuf[0]; __syncthreads();
  return r;
}

// ---------------- patchify + LN1 -> bf16[12544, 768] ----------------
__global__ __launch_bounds__(256)
void patchify_ln1_kernel(const float* __restrict__ img, const float* __restrict__ g,
                         const float* __restrict__ bt, __bf16* __restrict__ out) {
  __shared__ float sbuf[256];
  int tok = blockIdx.x;               // b*196 + n
  int b = tok / NTOK, n = tok % NTOK;
  int hh = n / 14, ww = n % 14;
  float vals[3]; float s = 0.f;
  #pragma unroll
  for (int r = 0; r < 3; r++) {
    int f = threadIdx.x + r * 256;    // feature = (p1*16+p2)*3 + c
    int c = f % 3, pp = f / 3, p2 = pp % 16, p1 = pp / 16;
    size_t idx = (((size_t)b * 3 + c) * 224 + (hh * 16 + p1)) * 224 + (ww * 16 + p2);
    vals[r] = img[idx]; s += vals[r];
  }
  float mean = block_sum(s, sbuf) * (1.0f / PD);
  float vs = 0.f;
  #pragma unroll
  for (int r = 0; r < 3; r++) { float d = vals[r] - mean; vs += d * d; }
  float inv = rsqrtf(block_sum(vs, sbuf) * (1.0f / PD) + LNEPS);
  #pragma unroll
  for (int r = 0; r < 3; r++) {
    int f = threadIdx.x + r * 256;
    out[(size_t)tok * PD + f] = (__bf16)((vals[r] - mean) * inv * g[f] + bt[f]);
  }
}

// ---------------- LN2(y)+pos -> padded f32 x[B,NPAD,DIM] ----------------
__global__ __launch_bounds__(256)
void ln2pos_kernel(const float* __restrict__ y, const float* __restrict__ g,
                   const float* __restrict__ bt, const float* __restrict__ pos,
                   float* __restrict__ x) {
  __shared__ float sbuf[256];
  int row = blockIdx.x;               // b*NPAD + n
  int b = row / NPAD, n = row % NPAD;
  float* xo = x + (size_t)row * DIMV;
  if (n >= NTOK) {
    #pragma unroll
    for (int r = 0; r < 4; r++) xo[threadIdx.x + r * 256] = 0.f;
    return;
  }
  const float* yi = y + (size_t)(b * NTOK + n) * DIMV;
  float vals[4]; float s = 0.f;
  #pragma unroll
  for (int r = 0; r < 4; r++) { vals[r] = yi[threadIdx.x + r * 256]; s += vals[r]; }
  float mean = block_sum(s, sbuf) * (1.0f / DIMV);
  float vs = 0.f;
  #pragma unroll
  for (int r = 0; r < 4; r++) { float d = vals[r] - mean; vs += d * d; }
  float inv = rsqrtf(block_sum(vs, sbuf) * (1.0f / DIMV) + LNEPS);
  #pragma unroll
  for (int r = 0; r < 4; r++) {
    int f = threadIdx.x + r * 256;
    xo[f] = (vals[r] - mean) * inv * g[f] + bt[f] + pos[(size_t)n * DIMV + f];
  }
}

// ---------------- LN(x) -> bf16 xn (zeroed pad rows) ----------------
__global__ __launch_bounds__(256)
void ln_bf16_kernel(const float* __restrict__ x, const float* __restrict__ g,
                    const float* __restrict__ bt, __bf16* __restrict__ xn) {
  __shared__ float sbuf[256];
  int row = blockIdx.x;
  int n = row % NPAD;
  __bf16* o = xn + (size_t)row * DIMV;
  if (n >= NTOK) {
    #pragma unroll
    for (int r = 0; r < 4; r++) o[threadIdx.x + r * 256] = (__bf16)0.f;
    return;
  }
  const float* xi = x + (size_t)row * DIMV;
  float vals[4]; float s = 0.f;
  #pragma unroll
  for (int r = 0; r < 4; r++) { vals[r] = xi[threadIdx.x + r * 256]; s += vals[r]; }
  float mean = block_sum(s, sbuf) * (1.0f / DIMV);
  float vs = 0.f;
  #pragma unroll
  for (int r = 0; r < 4; r++) { float d = vals[r] - mean; vs += d * d; }
  float inv = rsqrtf(block_sum(vs, sbuf) * (1.0f / DIMV) + LNEPS);
  #pragma unroll
  for (int r = 0; r < 4; r++) {
    int f = threadIdx.x + r * 256;
    o[f] = (__bf16)((vals[r] - mean) * inv * g[f] + bt[f]);
  }
}

// ---------------- masked softmax over 196 valid cols -> bf16 probs ----------------
__global__ __launch_bounds__(256)
void softmax_kernel(const float* __restrict__ S, __bf16* __restrict__ Ap) {
  __shared__ float sbuf[256];
  int row = blockIdx.x;               // b*NPAD + n
  int n = row % NPAD;
  int t = threadIdx.x;
  __bf16* o = Ap + (size_t)row * NPAD;
  if (n >= NTOK) { o[t] = (__bf16)0.f; return; }
  const float* si = S + (size_t)row * NPAD;
  float v = (t < NTOK) ? si[t] : -1e30f;
  float mx = block_max(v, sbuf);
  float e = (t < NTOK) ? __expf(v - mx) : 0.f;
  float ss = block_sum(e, sbuf);
  o[t] = (__bf16)(e / ss);
}

// ---------------- transpose f32[M,N] -> bf16[N,M] ----------------
__global__ __launch_bounds__(256)
void transpose_to_bf16_kernel(const float* __restrict__ src, __bf16* __restrict__ dst,
                              int M, int N) {
  size_t i = (size_t)blockIdx.x * 256 + threadIdx.x;   // index into dst [N,M]
  if (i >= (size_t)M * N) return;
  int m = (int)(i % M);
  int n = (int)(i / M);
  dst[i] = (__bf16)src[(size_t)m * N + n];
}

// ---------------- token mean ----------------
__global__ __launch_bounds__(256)
void mean_kernel(const float* __restrict__ x, float* __restrict__ xbar) {
  int b = blockIdx.x;
  for (int f = threadIdx.x; f < DIMV; f += 256) {
    float s = 0.f;
    for (int n = 0; n < NTOK; n++) s += x[((size_t)b * NPAD + n) * DIMV + f];
    xbar[(size_t)b * DIMV + f] = s * (1.0f / NTOK);
  }
}

// ---------------- classifier head (small, fp32) ----------------
__global__ __launch_bounds__(256)
void head_kernel(const float* __restrict__ xbar, const float* __restrict__ W,
                 const float* __restrict__ bias, float* __restrict__ out) {
  int i = blockIdx.x * 256 + threadIdx.x;
  if (i >= B_IMG * NCLS) return;
  int b = i / NCLS, o = i % NCLS;
  const float* xb = xbar + (size_t)b * DIMV;
  float s = bias[o];
  for (int k = 0; k < DIMV; k++) s += xb[k] * W[(size_t)k * NCLS + o];
  out[i] = s;
}

// ================= host launcher =================
extern "C" void kernel_launch(void* const* d_in, const int* in_sizes, int n_in,
                              void* d_out, int out_size, void* d_ws, size_t ws_size,
                              hipStream_t stream) {
  const float* image  = (const float*)d_in[0];
  const float* ln1_g  = (const float*)d_in[1];
  const float* ln1_b  = (const float*)d_in[2];
  const float* W_emb  = (const float*)d_in[3];
  const float* b_emb  = (const float*)d_in[4];
  const float* ln2_g  = (const float*)d_in[5];
  const float* ln2_b  = (const float*)d_in[6];
  const float* pos    = (const float*)d_in[7];
  const float* norm_g = (const float*)d_in[8];
  const float* norm_b = (const float*)d_in[9];
  const float* WV     = (const float*)d_in[10];
  const float* WK     = (const float*)d_in[11];
  const float* WQ     = (const float*)d_in[12];
  const float* lastW  = (const float*)d_in[13];
  const float* lastb  = (const float*)d_in[14];
  float* out = (float*)d_out;
  (void)in_sizes; (void)n_in; (void)out_size; (void)ws_size;

  char* wsp = (char*)d_ws;
  size_t off = 0;
  auto wsalloc = [&](size_t bytes) -> void* {
    void* p = wsp + off;
    off += (bytes + 255) & ~(size_t)255;
    return p;
  };

  __bf16* patch = (__bf16*)wsalloc((size_t)B_IMG * NTOK * PD * 2);
  __bf16* Wembt = (__bf16*)wsalloc((size_t)DIMV * PD * 2);
  __bf16* WQt   = (__bf16*)wsalloc((size_t)DIMV * DIMV * 2);
  __bf16* WKt   = (__bf16*)wsalloc((size_t)DIMV * DIMV * 2);
  __bf16* WVt   = (__bf16*)wsalloc((size_t)DHEAD * DHEAD * 2);
  float*  ybuf  = (float*) wsalloc((size_t)B_IMG * NTOK * DIMV * 4);
  float*  xbuf  = (float*) wsalloc((size_t)B_IMG * NPAD * DIMV * 4);
  __bf16* xn    = (__bf16*)wsalloc((size_t)B_IMG * NPAD * DIMV * 2);
  __bf16* qb    = (__bf16*)wsalloc((size_t)B_IMG * NPAD * DIMV * 2);
  __bf16* kb    = (__bf16*)wsalloc((size_t)B_IMG * NPAD * DIMV * 2);
  float*  Sbuf  = (float*) wsalloc((size_t)B_IMG * NPAD * NPAD * 4);
  __bf16* Ap    = (__bf16*)wsalloc((size_t)B_IMG * NPAD * NPAD * 2);
  __bf16* Gt    = (__bf16*)wsalloc((size_t)B_IMG * DIMV * NPAD * 2);
  float*  xbar  = (float*) wsalloc((size_t)B_IMG * DIMV * 4);

  const dim3 blk(256);
  const float scale = 0.08838834764831845f;  // 128^-0.5

  // --- weight prep (transpose + bf16) ---
  {
    size_t t;
    t = (size_t)DIMV * PD;
    transpose_to_bf16_kernel<<<dim3((unsigned)((t + 255) / 256)), blk, 0, stream>>>(W_emb, Wembt, PD, DIMV);
    t = (size_t)DIMV * DIMV;
    transpose_to_bf16_kernel<<<dim3((unsigned)((t + 255) / 256)), blk, 0, stream>>>(WQ, WQt, DIMV, DIMV);
    transpose_to_bf16_kernel<<<dim3((unsigned)((t + 255) / 256)), blk, 0, stream>>>(WK, WKt, DIMV, DIMV);
    t = (size_t)DHEAD * DHEAD;
    transpose_to_bf16_kernel<<<dim3((unsigned)((t + 255) / 256)), blk, 0, stream>>>(WV, WVt, DHEAD, DHEAD);
  }

  // --- patchify + LN1 ---
  patchify_ln1_kernel<<<dim3(B_IMG * NTOK), blk, 0, stream>>>(image, ln1_g, ln1_b, patch);

  // --- embed GEMM: y = patch @ W_emb + b_emb  (M=12544, N=1024, K=768) ---
  wmma_gemm_bf16<<<dim3(DIMV / BN, (B_IMG * NTOK) / BM, 1), blk, 0, stream>>>(
      patch, 0, PD, Wembt, 0, PD, ybuf, 0, DIMV,
      nullptr, 0, 0, b_emb, PD, 1.0f, OUT_F32);

  // --- LN2 + pos -> padded x ---
  ln2pos_kernel<<<dim3(B_IMG * NPAD), blk, 0, stream>>>(ybuf, ln2_g, ln2_b, pos, xbuf);

  for (int d = 0; d < DEPTH; d++) {
    // xn = LN(x), bf16, zero pad rows
    ln_bf16_kernel<<<dim3(B_IMG * NPAD), blk, 0, stream>>>(xbuf, norm_g, norm_b, xn);

    // q = xn @ WQ ; k = xn @ WK   (M=16384, N=1024, K=1024)
    wmma_gemm_bf16<<<dim3(DIMV / BN, (B_IMG * NPAD) / BM, 1), blk, 0, stream>>>(
        xn, 0, DIMV, WQt, 0, DIMV, qb, 0, DIMV,
        nullptr, 0, 0, nullptr, DIMV, 1.0f, OUT_BF16);
    wmma_gemm_bf16<<<dim3(DIMV / BN, (B_IMG * NPAD) / BM, 1), blk, 0, stream>>>(
        xn, 0, DIMV, WKt, 0, DIMV, kb, 0, DIMV,
        nullptr, 0, 0, nullptr, DIMV, 1.0f, OUT_BF16);

    // per-head G, stored transposed: Gt[b, h*128+e, n]  (M=256, N=128, K=128)
    for (int h = 0; h < NHEAD; h++) {
      wmma_gemm_bf16<<<dim3(DHEAD / BN, NPAD / BM, B_IMG), blk, 0, stream>>>(
          xn + (size_t)h * DHEAD, (long long)NPAD * DIMV, DIMV,
          WVt, 0, DHEAD,
          Gt + (size_t)h * DHEAD * NPAD, (long long)DIMV * NPAD, NPAD,
          nullptr, 0, 0, nullptr, DHEAD, 1.0f, OUT_BF16_T);
    }

    // S = scale * q @ k^T  (batched, M=N=256, K=1024)
    wmma_gemm_bf16<<<dim3(NPAD / BN, NPAD / BM, B_IMG), blk, 0, stream>>>(
        qb, (long long)NPAD * DIMV, DIMV,
        kb, (long long)NPAD * DIMV, DIMV,
        Sbuf, (long long)NPAD * NPAD, NPAD,
        nullptr, 0, 0, nullptr, DIMV, scale, OUT_F32);

    // A = softmax(S) masked to 196 cols, bf16 with zero pads
    softmax_kernel<<<dim3(B_IMG * NPAD), blk, 0, stream>>>(Sbuf, Ap);

    // x = A @ G + xn  (batched, M=256, N=1024, K=256)
    wmma_gemm_bf16<<<dim3(DIMV / BN, NPAD / BM, B_IMG), blk, 0, stream>>>(
        Ap, (long long)NPAD * NPAD, NPAD,
        Gt, (long long)DIMV * NPAD, NPAD,
        xbuf, (long long)NPAD * DIMV, DIMV,
        xn, (long long)NPAD * DIMV, DIMV,
        nullptr, NPAD, 1.0f, OUT_F32);
  }

  // mean over tokens, then classifier head
  mean_kernel<<<dim3(B_IMG), blk, 0, stream>>>(xbuf, xbar);
  head_kernel<<<dim3((B_IMG * NCLS + 255) / 256), blk, 0, stream>>>(xbar, lastW, lastb, out);
}
